// NinjaTurtleProjLinear_47304769798210
// MI455X (gfx1250) — matrix operand order repaired and checked by reading, not compiled, not run on hardware
//
#include <hip/hip_runtime.h>
#include <stdint.h>

// Problem dimensions (fixed by the reference).
#define M_TOK 8192   // tokens (rows of x / out)
#define KDIM  2048   // in_features
#define NOUT  8192   // out_features
#define NT16  (NOUT / 16)  // 512 N-tiles of width 16
#define KB32  (KDIM / 32)  // 64 K-blocks of depth 32

typedef __bf16 v16bf __attribute__((ext_vector_type(16)));
typedef float  v8f   __attribute__((ext_vector_type(8)));

struct alignas(16) U4 { unsigned x, y, z, w; };
struct U8 { U4 lo, hi; };  // 32 bytes == sizeof(v16bf)

// ---------- bf16 split helpers (round-to-nearest-even) ----------
__device__ __forceinline__ unsigned short f2bf(float f) {
  union { float f; unsigned u; } v; v.f = f;
  unsigned u = v.u;
  u += 0x7FFFu + ((u >> 16) & 1u);
  return (unsigned short)(u >> 16);
}
__device__ __forceinline__ float bf2f(unsigned short h) {
  union { unsigned u; float f; } v; v.u = ((unsigned)h) << 16;
  return v.f;
}

// ---------- prep: split x into bf16 hi/lo ----------
__global__ void pack_x_kernel(const float* __restrict__ x,
                              unsigned short* __restrict__ xh,
                              unsigned short* __restrict__ xl) {
  size_t i = (size_t)blockIdx.x * blockDim.x + threadIdx.x;
  float v = x[i];
  unsigned short h = f2bf(v);
  xh[i] = h;
  xl[i] = f2bf(v - bf2f(h));
}

// ---------- prep: fold mask into W, split hi/lo, mark nonzero blocks ----------
__global__ void pack_w_kernel(const float* __restrict__ w,
                              const float* __restrict__ mask,
                              unsigned short* __restrict__ wh,
                              unsigned short* __restrict__ wl,
                              unsigned* __restrict__ flags) {
  size_t i = (size_t)blockIdx.x * blockDim.x + threadIdx.x;
  float mv = mask[i];
  float v = mv * w[i];
  unsigned short h = f2bf(v);
  wh[i] = h;
  wl[i] = f2bf(v - bf2f(h));
  if (mv != 0.0f) {
    int n = (int)(i / KDIM);
    int k = (int)(i % KDIM);
    atomicOr(&flags[(n >> 4) * KB32 + (k >> 5)], 1u);
  }
}

// ---------- prep: compact flags into per-N-tile K-block lists ----------
__global__ void build_klist_kernel(const unsigned* __restrict__ flags,
                                   unsigned short* __restrict__ klist,
                                   unsigned* __restrict__ counts) {
  int t = blockIdx.x * blockDim.x + threadIdx.x;
  if (t >= NT16) return;
  int c = 0;
  for (int kb = 0; kb < KB32; ++kb)
    if (flags[t * KB32 + kb]) klist[t * KB32 + c++] = (unsigned short)kb;
  counts[t] = (unsigned)c;
}

// ---------- fragment loads straight from row-major global memory ----------
// A (16x32 bf16, MxK): lane m<16 holds M=m, K chunks [0..7] and [16..23];
// lane m>=16 holds M=m-16, K chunks [8..15] and [24..31]. Caller pre-adds mh*8.
__device__ __forceinline__ v16bf load_frag_a(const unsigned short* p) {
  U8 t;
  t.lo = *(const U4*)p;          // 8 bf16: K base .. base+7
  t.hi = *(const U4*)(p + 16);   // 8 bf16: K base+16 .. base+23
  return __builtin_bit_cast(v16bf, t);
}
// B (32x16 bf16, KxN): lane n<16 holds N=n, K chunks [0..7] and [16..23];
// lane n>=16 holds N=n-16, K chunks [8..15] and [24..31]. Caller pre-adds mh*8.
__device__ __forceinline__ v16bf load_frag_b(const unsigned short* p) {
  U8 t;
  t.lo = *(const U4*)p;          // K base .. base+7
  t.hi = *(const U4*)(p + 16);   // K base+16 .. base+23
  return __builtin_bit_cast(v16bf, t);
}

// ---------- block-sparse split-bf16 WMMA GEMM ----------
// grid = (NT16, M_TOK/16/8); 256 threads = 8 waves; wave w -> m_tile,
// whole block shares one n_tile (shared B fragments hit in L0/L2).
__global__ void __launch_bounds__(256)
sparse_wmma_gemm(const unsigned short* __restrict__ xh,
                 const unsigned short* __restrict__ xl,
                 const unsigned short* __restrict__ wh,
                 const unsigned short* __restrict__ wl,
                 const unsigned short* __restrict__ klist,
                 const unsigned* __restrict__ counts,
                 const float* __restrict__ bias,
                 float* __restrict__ out) {
  const int lane = (int)(threadIdx.x & 31u);
  const int wave = (int)(threadIdx.x >> 5u);
  const int n_tile = (int)blockIdx.x;
  const int m_tile = (int)blockIdx.y * 8 + wave;
  const int lr = lane & 15;   // row/col within tile
  const int mh = lane >> 4;   // half-wave select
  const int n0 = n_tile << 4;
  const int m0 = m_tile << 4;

  const unsigned short* xh_row = xh + (size_t)(m0 + lr) * KDIM + mh * 8;
  const unsigned short* xl_row = xl + (size_t)(m0 + lr) * KDIM + mh * 8;
  const unsigned short* wh_row = wh + (size_t)(n0 + lr) * KDIM + mh * 8;
  const unsigned short* wl_row = wl + (size_t)(n0 + lr) * KDIM + mh * 8;
  const unsigned short* kl = klist + n_tile * KB32;
  const int cnt = (int)counts[n_tile];

  v8f acc = {};
  for (int i = 0; i < cnt; ++i) {
    const int k0 = ((int)kl[i]) << 5;
    v16bf ah = load_frag_a(xh_row + k0);
    v16bf al = load_frag_a(xl_row + k0);
    v16bf bh = load_frag_b(wh_row + k0);
    v16bf bl = load_frag_b(wl_row + k0);
    // fp32 via 3-term bf16 split: hi*hi + hi*lo + lo*hi
    acc = __builtin_amdgcn_wmma_f32_16x16x32_bf16(false, ah, false, bh,
                                                  (short)0, acc, false, false);
    acc = __builtin_amdgcn_wmma_f32_16x16x32_bf16(false, ah, false, bl,
                                                  (short)0, acc, false, false);
    acc = __builtin_amdgcn_wmma_f32_16x16x32_bf16(false, al, false, bh,
                                                  (short)0, acc, false, false);
  }

  // C/D layout: VGPR r -> M = r + 8*mh, lane -> N = n0 + lr.
  const float bv = bias[n0 + lr];
  float* orow = out + (size_t)(m0 + mh * 8) * NOUT + (n0 + lr);
#pragma unroll
  for (int r = 0; r < 8; ++r) orow[(size_t)r * NOUT] = acc[r] + bv;
}

// ---------- safety fallback if workspace is too small ----------
__global__ void fallback_kernel(const float* __restrict__ x,
                                const float* __restrict__ w,
                                const float* __restrict__ b,
                                const float* __restrict__ mask,
                                float* __restrict__ out) {
  size_t idx = (size_t)blockIdx.x * blockDim.x + threadIdx.x;
  const int col = (int)(idx % NOUT);
  const size_t row = idx / NOUT;
  const float* xr = x + row * (size_t)KDIM;
  const float* wr = w + (size_t)col * KDIM;
  const float* mr = mask + (size_t)col * KDIM;
  float s = 0.0f;
  for (int k = 0; k < KDIM; ++k) s += xr[k] * (mr[k] * wr[k]);
  out[idx] = s + b[col];
}

extern "C" void kernel_launch(void* const* d_in, const int* in_sizes, int n_in,
                              void* d_out, int out_size, void* d_ws, size_t ws_size,
                              hipStream_t stream) {
  (void)in_sizes; (void)n_in; (void)out_size;
  const float* x    = (const float*)d_in[0];
  const float* w    = (const float*)d_in[1];
  const float* bias = (const float*)d_in[2];
  const float* mask = (const float*)d_in[3];
  float* out = (float*)d_out;

  const size_t SZX = (size_t)M_TOK * KDIM;  // 16,777,216 elements
  const size_t SZW = (size_t)NOUT * KDIM;   // 16,777,216 elements

  char* ws = (char*)d_ws;
  size_t off = 0;
  unsigned short* xh = (unsigned short*)(ws + off); off += SZX * 2;
  unsigned short* xl = (unsigned short*)(ws + off); off += SZX * 2;
  unsigned short* wh = (unsigned short*)(ws + off); off += SZW * 2;
  unsigned short* wl = (unsigned short*)(ws + off); off += SZW * 2;
  unsigned*    flags = (unsigned*)(ws + off);       off += (size_t)NT16 * KB32 * 4;
  unsigned short* kl = (unsigned short*)(ws + off); off += (size_t)NT16 * KB32 * 2;
  unsigned*   counts = (unsigned*)(ws + off);       off += (size_t)NT16 * 4;

  if (ws_size < off) {
    const size_t total = (size_t)M_TOK * NOUT;
    fallback_kernel<<<(unsigned)(total / 256), 256, 0, stream>>>(x, w, bias, mask, out);
    return;
  }

  hipMemsetAsync(flags, 0, (size_t)NT16 * KB32 * 4, stream);
  pack_x_kernel<<<(unsigned)(SZX / 256), 256, 0, stream>>>(x, xh, xl);
  pack_w_kernel<<<(unsigned)(SZW / 256), 256, 0, stream>>>(w, mask, wh, wl, flags);
  build_klist_kernel<<<(NT16 + 255) / 256, 256, 0, stream>>>(flags, kl, counts);

  dim3 grid(NT16, M_TOK / 16 / 8);  // 512 x 64 blocks, 8 waves each
  sparse_wmma_gemm<<<grid, 256, 0, stream>>>(xh, xl, wh, wl, kl, counts, bias, out);
}
